// CacheAttention_1486058685085
// MI455X (gfx1250) — compile-verified, matrix operand order checked by reading, not run on hardware
//
#include <hip/hip_runtime.h>
#include <stdint.h>

typedef __bf16 v16bf __attribute__((ext_vector_type(16)));
typedef float  v8f   __attribute__((ext_vector_type(8)));

#define B_     2
#define S_     1024
#define NH_    16
#define HD_    128
#define NN_    16
#define NMAT_  2048                      // NH*HD
#define SCALE_ 0.08838834764831845f     // 128^-0.5

// ---------------- helpers ----------------
__device__ __forceinline__ float bf2f(uint32_t bits16) {
  union { uint32_t u; float f; } v; v.u = bits16 << 16; return v.f;
}
struct __align__(8) bf16x4 { __bf16 x, y, z, w; };

// ---------------- f32 -> bf16 elementwise ----------------
__global__ __launch_bounds__(256) void cvt_bf16_kernel(const float* __restrict__ in,
                                                       __bf16* __restrict__ out, long n4) {
  long i = (long)blockIdx.x * 256 + threadIdx.x;
  if (i >= n4) return;
  float4 f = ((const float4*)in)[i];
  bf16x4 o; o.x = (__bf16)f.x; o.y = (__bf16)f.y; o.z = (__bf16)f.z; o.w = (__bf16)f.w;
  ((bf16x4*)out)[i] = o;
}

// ---------------- W (K x N) f32  ->  WT (N x K) bf16 ----------------
__global__ __launch_bounds__(256) void transpose_cvt_kernel(const float* __restrict__ W,
                                                            __bf16* __restrict__ WT, int K, int N) {
  __shared__ float t[32][33];
  int tx = threadIdx.x & 31, ty = threadIdx.x >> 5;
  int k0 = blockIdx.y * 32, n0 = blockIdx.x * 32;
#pragma unroll
  for (int i = 0; i < 4; ++i)
    t[ty + i * 8][tx] = W[(size_t)(k0 + ty + i * 8) * N + n0 + tx];
  __syncthreads();
#pragma unroll
  for (int i = 0; i < 4; ++i)
    WT[(size_t)(n0 + ty + i * 8) * K + k0 + tx] = (__bf16)t[tx][ty + i * 8];
}

// ---------------- bf16 WMMA GEMM ----------------
// C[M,N] = alpha * A[M,K] @ BT[N,K]^T (+resid).  DUAL: second B/second C (both bf16 out).
#define BM  128
#define BN  128
#define BK  32
#define LDT 40            // bf16 elems per LDS row: 32 + 8 pad (80 bytes, conflict-free b128)
#define TILE_BYTES (BM * LDT * 2)

union Frag { v16bf v; uint4 q[2]; };

__device__ __forceinline__ void async_stage_tile(const __bf16* __restrict__ g, int ld,
                                                 long row0, int k0, uint32_t lds_base, int tid) {
#pragma unroll
  for (int i = 0; i < 2; ++i) {
    int c = tid + i * 256;
    int r = c >> 2, ch = c & 3;                       // 128 rows x 4 chunks of 16B
    const __bf16* gp = g + (size_t)(row0 + r) * ld + (k0 + ch * 8);
    uint32_t loff = lds_base + (uint32_t)(r * (LDT * 2) + ch * 16);
    asm volatile("global_load_async_to_lds_b128 %0, %1, off" :: "v"(loff), "v"(gp) : "memory");
  }
}

__device__ __forceinline__ Frag load_frag(const __bf16* base, int row, int half) {
  Frag f;
  f.q[0] = *(const uint4*)(base + row * LDT + half * 8);       // K = kbase .. kbase+7
  f.q[1] = *(const uint4*)(base + row * LDT + 16 + half * 8);  // K = 16+kbase .. +7
  return f;
}

template <bool DUAL, bool OUTBF16, bool RES>
__global__ __launch_bounds__(256) void gemm_bf16_wmma(
    const __bf16* __restrict__ A, const __bf16* __restrict__ BT0, const __bf16* __restrict__ BT1,
    void* __restrict__ C0, void* __restrict__ C1, const float* __restrict__ resid,
    int N, int K, float alpha) {
  __shared__ __align__(16) __bf16 sA [2][BM * LDT];
  __shared__ __align__(16) __bf16 sB0[2][BN * LDT];
  __shared__ __align__(16) __bf16 sB1[2][BN * LDT];

  const int tid = threadIdx.x;
  const int lane = tid & 31;
  const int wid = tid >> 5;
  const int wave_m = wid & 3;     // 4 slabs of 32 rows
  const int wave_n = wid >> 2;    // 2 slabs of 64 cols
  const long row0 = (long)blockIdx.y * BM;
  const int  col0 = blockIdx.x * BN;
  const int mlane = lane & 15;
  const int half  = lane >> 4;

  v8f acc0[2][4], acc1[2][4];
#pragma unroll
  for (int i = 0; i < 2; ++i)
#pragma unroll
    for (int j = 0; j < 4; ++j)
#pragma unroll
      for (int r = 0; r < 8; ++r) { acc0[i][j][r] = 0.f; if (DUAL) acc1[i][j][r] = 0.f; }

  const uint32_t aBase  = (uint32_t)(uintptr_t)&sA[0][0];
  const uint32_t b0Base = (uint32_t)(uintptr_t)&sB0[0][0];
  const uint32_t b1Base = (uint32_t)(uintptr_t)&sB1[0][0];

  const int nk = K / BK;
  // prologue: stage k-block 0 into buffer 0
  async_stage_tile(A,   K, row0, 0, aBase,  tid);
  async_stage_tile(BT0, K, col0, 0, b0Base, tid);
  if (DUAL) async_stage_tile(BT1, K, col0, 0, b1Base, tid);

  for (int kb = 0; kb < nk; ++kb) {
    const int cur = kb & 1;
    asm volatile("s_wait_asynccnt 0x0" ::: "memory");   // own async copies done
    __syncthreads();                                    // everyone's copies visible; prev reads done
    if (kb + 1 < nk) {
      const int nxt = cur ^ 1;
      const int k1 = (kb + 1) * BK;
      async_stage_tile(A,   K, row0, k1, aBase  + (uint32_t)nxt * TILE_BYTES, tid);
      async_stage_tile(BT0, K, col0, k1, b0Base + (uint32_t)nxt * TILE_BYTES, tid);
      if (DUAL) async_stage_tile(BT1, K, col0, k1, b1Base + (uint32_t)nxt * TILE_BYTES, tid);
    }
    const __bf16* sa  = &sA[cur][0];
    const __bf16* sb0 = &sB0[cur][0];

    // batch ALL fragment loads for this k-block into distinct registers, then
    // issue the WMMA burst back-to-back (lets dscnt drain progressively
    // instead of fencing each WMMA pair on s_wait_dscnt 0).
    Frag a0 = load_frag(sa, wave_m * 32 + 0 * 16 + mlane, half);
    Frag a1 = load_frag(sa, wave_m * 32 + 1 * 16 + mlane, half);
    Frag b[4];
#pragma unroll
    for (int tn = 0; tn < 4; ++tn)
      b[tn] = load_frag(sb0, wave_n * 64 + tn * 16 + mlane, half);
#pragma unroll
    for (int tn = 0; tn < 4; ++tn) {
      acc0[0][tn] = __builtin_amdgcn_wmma_f32_16x16x32_bf16(false, a0.v, false, b[tn].v, (short)0, acc0[0][tn], false, false);
      acc0[1][tn] = __builtin_amdgcn_wmma_f32_16x16x32_bf16(false, a1.v, false, b[tn].v, (short)0, acc0[1][tn], false, false);
    }
    if (DUAL) {
      const __bf16* sb1 = &sB1[cur][0];
      Frag c[4];
#pragma unroll
      for (int tn = 0; tn < 4; ++tn)
        c[tn] = load_frag(sb1, wave_n * 64 + tn * 16 + mlane, half);
#pragma unroll
      for (int tn = 0; tn < 4; ++tn) {
        acc1[0][tn] = __builtin_amdgcn_wmma_f32_16x16x32_bf16(false, a0.v, false, c[tn].v, (short)0, acc1[0][tn], false, false);
        acc1[1][tn] = __builtin_amdgcn_wmma_f32_16x16x32_bf16(false, a1.v, false, c[tn].v, (short)0, acc1[1][tn], false, false);
      }
    }
  }

  // writeback: C/D layout -> lane n = lane&15, rows m = (lane>>4)*8 + r
  const long mbase = row0 + wave_m * 32 + half * 8;
  const int  nbase = col0 + wave_n * 64;
#pragma unroll
  for (int tm = 0; tm < 2; ++tm)
#pragma unroll
    for (int tn = 0; tn < 4; ++tn) {
      const long mrow = mbase + tm * 16;
      const int  ncol = nbase + tn * 16 + mlane;
#pragma unroll
      for (int r = 0; r < 8; ++r) {
        const long idx = (mrow + r) * (long)N + ncol;
        float val = acc0[tm][tn][r] * alpha;
        if (RES) val += resid[idx];
        if (OUTBF16) ((__bf16*)C0)[idx] = (__bf16)val;
        else         ((float*)C0)[idx] = val;
        if (DUAL)    ((__bf16*)C1)[idx] = (__bf16)(acc1[tm][tn][r] * alpha);
      }
    }
}

// ---------------- local attention: one wave per (b,q,h) ----------------
__global__ __launch_bounds__(256) void attn_kernel(const float* __restrict__ Q,
                                                   const __bf16* __restrict__ Kb,
                                                   const __bf16* __restrict__ Vb,
                                                   __bf16* __restrict__ Aout) {
  const int w = blockIdx.x * 8 + (threadIdx.x >> 5);
  const int lane = threadIdx.x & 31;
  const int b = w >> 14;              // / (S*NH)
  const int rem = w & 16383;
  const int q = rem >> 4;
  const int h = rem & 15;

  // score for key i = lane: keys 0..15 from pos q-1 (clipped), 16..31 from pos q
  const int posi = (lane < 16) ? (q > 0 ? q - 1 : 0) : q;
  const int il = lane & 15;
  const float* qp = Q + ((size_t)(b * S_ + q)) * NMAT_ + h * HD_;
  const uint32_t* kp = (const uint32_t*)(Kb + (((size_t)(b * S_ + posi)) * NN_ + h) * NMAT_ + il * HD_);

  float s = 0.f;
#pragma unroll 16
  for (int d2 = 0; d2 < HD_ / 2; ++d2) {
    uint32_t u = kp[d2];
    s += qp[2 * d2]     * bf2f(u & 0xffffu);
    s += qp[2 * d2 + 1] * bf2f(u >> 16);
  }
  float m = s;
#pragma unroll
  for (int off = 16; off >= 1; off >>= 1) m = fmaxf(m, __shfl_xor(m, off, 32));
  float e = __expf(s - m);
  float sum = e;
#pragma unroll
  for (int off = 16; off >= 1; off >>= 1) sum += __shfl_xor(sum, off, 32);
  const float p = e / sum;

  const int d0 = lane * 4;            // each lane produces 4 of the 128 outputs
  float a0 = 0.f, a1 = 0.f, a2 = 0.f, a3 = 0.f;
#pragma unroll 8
  for (int j = 0; j < 32; ++j) {
    float pj = __shfl(p, j, 32);
    int pos = (j < 16) ? (q > 0 ? q - 1 : 0) : q;
    int jl = j & 15;
    const uint32_t* vp = (const uint32_t*)(Vb + (((size_t)(b * S_ + pos)) * NN_ + h) * NMAT_ + jl * HD_ + d0);
    uint32_t u0 = vp[0], u1 = vp[1];
    a0 += pj * bf2f(u0 & 0xffffu);
    a1 += pj * bf2f(u0 >> 16);
    a2 += pj * bf2f(u1 & 0xffffu);
    a3 += pj * bf2f(u1 >> 16);
  }
  bf16x4 o; o.x = (__bf16)a0; o.y = (__bf16)a1; o.z = (__bf16)a2; o.w = (__bf16)a3;
  *(bf16x4*)(Aout + ((size_t)(b * S_ + q)) * NMAT_ + h * HD_ + d0) = o;
}

// ---------------- launcher ----------------
extern "C" void kernel_launch(void* const* d_in, const int* in_sizes, int n_in,
                              void* d_out, int out_size, void* d_ws, size_t ws_size,
                              hipStream_t stream) {
  (void)in_sizes; (void)n_in; (void)out_size; (void)ws_size;
  const float* hs  = (const float*)d_in[0];
  const float* ext = (const float*)d_in[1];
  const float* Wq  = (const float*)d_in[2];
  const float* Wk  = (const float*)d_in[3];
  const float* Wv  = (const float*)d_in[4];
  const float* Wo  = (const float*)d_in[5];
  float* out = (float*)d_out;

  char* ws = (char*)d_ws; size_t off = 0;
  auto carve = [&](size_t bytes) { void* p = ws + off; off += (bytes + 255) & ~(size_t)255; return p; };
  __bf16* Xbf = (__bf16*)carve((size_t)2048 * 2048 * 2);   // hidden_states bf16
  __bf16* Ebf = (__bf16*)carve((size_t)32768 * 1024 * 2);  // ext_hidden_states bf16
  __bf16* WqT = (__bf16*)carve((size_t)2048 * 2048 * 2);   // N x K
  __bf16* WkT = (__bf16*)carve((size_t)2048 * 1024 * 2);
  __bf16* WvT = (__bf16*)carve((size_t)2048 * 1024 * 2);
  __bf16* WoT = (__bf16*)carve((size_t)2048 * 2048 * 2);
  float*  Qf  = (float*) carve((size_t)2048 * 2048 * 4);   // Q (scaled), f32
  __bf16* Kbf = (__bf16*)carve((size_t)32768 * 2048 * 2);
  __bf16* Vbf = (__bf16*)carve((size_t)32768 * 2048 * 2);
  __bf16* Abf = (__bf16*)carve((size_t)2048 * 2048 * 2);   // attention output

  cvt_bf16_kernel<<<4096,  256, 0, stream>>>(hs,  Xbf, (long)(2048L * 2048 / 4));
  cvt_bf16_kernel<<<32768, 256, 0, stream>>>(ext, Ebf, (long)(32768L * 1024 / 4));

  transpose_cvt_kernel<<<dim3(64, 64), 256, 0, stream>>>(Wq, WqT, 2048, 2048);
  transpose_cvt_kernel<<<dim3(64, 32), 256, 0, stream>>>(Wk, WkT, 1024, 2048);
  transpose_cvt_kernel<<<dim3(64, 32), 256, 0, stream>>>(Wv, WvT, 1024, 2048);
  transpose_cvt_kernel<<<dim3(64, 64), 256, 0, stream>>>(Wo, WoT, 2048, 2048);

  // Q = X @ Wq * scale  (f32 out)
  gemm_bf16_wmma<false, false, false><<<dim3(16, 16), 256, 0, stream>>>(
      Xbf, WqT, nullptr, Qf, nullptr, nullptr, 2048, 2048, SCALE_);
  // K,V = E @ Wk / E @ Wv  (fused, bf16 out)
  gemm_bf16_wmma<true, true, false><<<dim3(16, 256), 256, 0, stream>>>(
      Ebf, WkT, WvT, Kbf, Vbf, nullptr, 2048, 1024, 1.0f);
  // local softmax attention
  attn_kernel<<<4096, 256, 0, stream>>>(Qf, Kbf, Vbf, Abf);
  // out = A @ Wo + hidden_states  (f32 out, residual)
  gemm_bf16_wmma<false, false, true><<<dim3(16, 16), 256, 0, stream>>>(
      Abf, WoT, nullptr, out, nullptr, hs, 2048, 2048, 1.0f);
}
